// DecoderLSTM_78494822301856
// MI455X (gfx1250) — matrix operand order repaired and checked by reading, not compiled, not run on hardware
//
#include <hip/hip_runtime.h>
#include <hip/hip_bf16.h>

// Problem geometry (fixed by setup_inputs of the reference)
#define HID    512
#define EMB    512
#define VOCAB  1000
#define NPAD   1008          // VOCAB padded to multiple of 16
#define NT_OUT 63            // NPAD/16 logits tiles
#define BATCH  128
#define TSTEPS 256           // max_seq_len (loop count must be host-static)
#define G4H    2048          // 4*HID
#define KCAT   1024          // EMB + HID

// Persistent-kernel partitioning: 8 independent groups x 4 workgroups,
// each group owns 16 batch rows for the whole 256-step decode.
#define NGROUPS        8
#define WG_PER_GROUP   4
#define ROWS_PER_GROUP 16
#define WAVES_PER_GROUP 32   // WG_PER_GROUP * 8 waves

typedef __attribute__((ext_vector_type(16))) _Float16 v16h;
typedef __attribute__((ext_vector_type(8)))  _Float16 v8h;
typedef __attribute__((ext_vector_type(8)))  float    v8f;

__device__ __forceinline__ float sigmoid_f(float x) { return 1.0f / (1.0f + __expf(-x)); }

// ---- WMMA fragment helpers (CDNA5 ISA 7.12.2 layouts, wave32) -------------
// A (16-bit 16x32): lanes 0-15 -> K {k0..k0+7, k0+16..k0+23} of row lane,
//                   lanes 16-31 -> K {k0+8..k0+15, k0+24..k0+31}
__device__ __forceinline__ v16h load_a_frag(const _Float16* Arow, int k0, int hi)
{
    const int aoff = k0 + (hi ? 8 : 0);
    v8h alo = *(const v8h*)(Arow + aoff);
    v8h ahi = *(const v8h*)(Arow + aoff + 16);
    return __builtin_shufflevector(alo, ahi, 0,1,2,3,4,5,6,7,8,9,10,11,12,13,14,15);
}
// B (16-bit 32x16): lanes 0-15 -> K {k0..k0+7, k0+16..k0+23} of column lane,
//                   lanes 16-31 -> K {k0+8..k0+15, k0+24..k0+31}
__device__ __forceinline__ v16h load_b_frag(const _Float16* Brow, int k0, int hi)
{
    return __builtin_shufflevector(*(const v8h*)(Brow + k0 + (hi ? 8 : 0)), *(const v8h*)(Brow + k0 + (hi ? 8 : 0) + 16), 0,1,2,3,4,5,6,7,8,9,10,11,12,13,14,15);
}
__device__ __forceinline__ v8f wmma_acc(v16h a, v16h b, v8f c)
{
    return __builtin_amdgcn_wmma_f32_16x16x32_f16(false, a, false, b, (short)0, c, false, false);
}

// Order-preserving float->uint map (larger float -> larger uint)
__device__ __forceinline__ unsigned f2ord(float f)
{
    unsigned u = __float_as_uint(f);
    return (u & 0x80000000u) ? ~u : (u | 0x80000000u);
}

__device__ __forceinline__ unsigned long long shflxor_u64(unsigned long long v, int msk, int w)
{
    unsigned lo = (unsigned)v, hh = (unsigned)(v >> 32);
    lo = (unsigned)__shfl_xor((int)lo, msk, w);
    hh = (unsigned)__shfl_xor((int)hh, msk, w);
    return ((unsigned long long)hh << 32) | lo;
}

// Monotonic counter barrier across the 4 workgroups of one group.
__device__ __forceinline__ void group_barrier(unsigned* ctr, unsigned target)
{
    __syncthreads();
    if (threadIdx.x == 0) {
        __threadfence();  // release
        __hip_atomic_fetch_add(ctr, 1u, __ATOMIC_RELAXED, __HIP_MEMORY_SCOPE_AGENT);
        while (__hip_atomic_load(ctr, __ATOMIC_RELAXED, __HIP_MEMORY_SCOPE_AGENT) < target)
            __builtin_amdgcn_s_sleep(1);
        __threadfence();  // acquire
    }
    __syncthreads();
}

// ---------------------------------------------------------------------------
// One-time weight prep (f32 -> f16 with concat / pad), small kernels
// ---------------------------------------------------------------------------
__global__ void conv_wcat(const float* __restrict__ W_ih, const float* __restrict__ W_hh,
                          _Float16* __restrict__ Wcat)
{
    int idx = blockIdx.x * blockDim.x + threadIdx.x;      // G4H*KCAT
    int nrow = idx >> 10, k = idx & 1023;
    float v = (k < HID) ? W_ih[(size_t)nrow * KCAT + HID + k]
                        : W_hh[(size_t)nrow * HID + (k - HID)];
    Wcat[idx] = (_Float16)v;
}
__global__ void conv_wctx(const float* __restrict__ W_ih, _Float16* __restrict__ Wctx)
{
    int idx = blockIdx.x * blockDim.x + threadIdx.x;      // G4H*HID
    int nrow = idx >> 9, k = idx & 511;
    Wctx[idx] = (_Float16)W_ih[(size_t)nrow * KCAT + k];
}
__global__ void conv_wout(const float* __restrict__ W_out, _Float16* __restrict__ Wout16)
{
    int idx = blockIdx.x * blockDim.x + threadIdx.x;      // NPAD*HID
    int nrow = idx >> 9;
    Wout16[idx] = (_Float16)((nrow < VOCAB) ? W_out[idx] : 0.0f);
}
__global__ void conv_bsum(const float* __restrict__ b_ih, const float* __restrict__ b_hh,
                          float* __restrict__ bsum, unsigned* __restrict__ bar)
{
    int idx = blockIdx.x * blockDim.x + threadIdx.x;      // G4H
    bsum[idx] = b_ih[idx] + b_hh[idx];
    if (idx < NGROUPS) bar[idx] = 0u;                     // reset barriers every call
}
__global__ void init_state(const float* __restrict__ context,
                           const float* __restrict__ embed,
                           const int* __restrict__ start_id,
                           _Float16* __restrict__ xcat0, _Float16* __restrict__ ctx16,
                           float* __restrict__ h, float* __restrict__ c)
{
    int idx = blockIdx.x * blockDim.x + threadIdx.x;      // BATCH*HID
    int b = idx >> 9, j = idx & 511;
    float cv = context[idx];
    h[idx] = cv;
    c[idx] = 0.0f;
    ctx16[idx] = (_Float16)cv;
    xcat0[(size_t)b * KCAT + HID + j] = (_Float16)cv;
    xcat0[(size_t)b * KCAT + j] = (_Float16)embed[(size_t)(*start_id) * EMB + j];
}

// ---------------------------------------------------------------------------
// Persistent decoder: one launch runs all 256 steps.
// 32 blocks x 256 threads = 8 groups x (4 WGs = 32 waves), 16 rows per group.
// ---------------------------------------------------------------------------
__global__ __launch_bounds__(256) void decoder_persistent(
    const _Float16* __restrict__ Wcat,   // [2048, 1024]  f16 ( [W_ih_emb | W_hh] )
    const _Float16* __restrict__ Wctx,   // [2048, 512]   f16 ( W_ih context part )
    const _Float16* __restrict__ Wout16, // [1008, 512]   f16, zero padded
    const float*    __restrict__ bsum,   // [2048]  b_ih + b_hh
    const float*    __restrict__ b_out,  // [1000]
    const float*    __restrict__ embed,  // [1000, 512] f32
    const _Float16* __restrict__ ctx16,  // [128, 512] f16 context
    float*          __restrict__ ctxg,   // [128, 2048] f32 scratch
    _Float16*       __restrict__ xcat0,  // [128, 1024] f16 (double buffered)
    _Float16*       __restrict__ xcat1,
    float*          __restrict__ hbuf,   // [128, 512] f32
    float*          __restrict__ cbuf,   // [128, 512] f32
    unsigned long long* __restrict__ part, // [128, 64] packed argmax partials
    unsigned*       __restrict__ bar,    // [NGROUPS] barrier counters
    float*          __restrict__ out)    // [128,256,1000] ++ h ++ c
{
    const int group  = blockIdx.x / WG_PER_GROUP;
    const int wgIn   = blockIdx.x % WG_PER_GROUP;
    const int waveIn = wgIn * 8 + (int)(threadIdx.x >> 5);
    const int lane   = (int)(threadIdx.x & 31);
    const int hi = lane >> 4;
    const int lq = lane & 15;
    const int m0 = group * ROWS_PER_GROUP;
    unsigned* ctr = bar + group;
    unsigned phase = 0;

    // ---- ctx_gates = context @ Wctx^T + (b_ih + b_hh)   [16 x 2048] per group
    {
        const _Float16* Arow = ctx16 + (size_t)(m0 + lq) * HID;
        for (int tn = waveIn; tn < G4H / 16; tn += WAVES_PER_GROUP) {
            const _Float16* Brow = Wctx + (size_t)(tn * 16 + lq) * HID;
            v8f acc = {};
            for (int k0 = 0; k0 < HID; k0 += 32)
                acc = wmma_acc(load_a_frag(Arow, k0, hi), load_b_frag(Brow, k0, hi), acc);
            const int n = tn * 16 + lq;
            const float bv = bsum[n];
#pragma unroll
            for (int r = 0; r < 8; ++r)
                ctxg[(size_t)(m0 + hi * 8 + r) * G4H + n] = acc[r] + bv;
        }
    }
    group_barrier(ctr, ++phase * WG_PER_GROUP);

    const size_t row_stride_out = (size_t)TSTEPS * VOCAB;

    for (int t = 0; t < TSTEPS; ++t) {
        _Float16* xc = (t & 1) ? xcat1 : xcat0;   // current [emb|h]
        _Float16* xn = (t & 1) ? xcat0 : xcat1;   // next    [emb|h]

        // ---- Phase B: gates GEMM (i,f,g,o tiles in one wave) + LSTM cell ----
        {
            const int jt = waveIn;                // 32 waves <-> 32 j-tiles
            const int j  = jt * 16 + lq;
            const _Float16* Arow = xc + (size_t)(m0 + lq) * KCAT;
            const _Float16* Br0 = Wcat + (size_t)(0 * HID + j) * KCAT;
            const _Float16* Br1 = Wcat + (size_t)(1 * HID + j) * KCAT;
            const _Float16* Br2 = Wcat + (size_t)(2 * HID + j) * KCAT;
            const _Float16* Br3 = Wcat + (size_t)(3 * HID + j) * KCAT;
            v8f ai = {}, af = {}, ag = {}, ao = {};
            for (int k0 = 0; k0 < KCAT; k0 += 32) {
                v16h a = load_a_frag(Arow, k0, hi);        // reused 4x
                ai = wmma_acc(a, load_b_frag(Br0, k0, hi), ai);
                af = wmma_acc(a, load_b_frag(Br1, k0, hi), af);
                ag = wmma_acc(a, load_b_frag(Br2, k0, hi), ag);
                ao = wmma_acc(a, load_b_frag(Br3, k0, hi), ao);
            }
#pragma unroll
            for (int r = 0; r < 8; ++r) {
                const int m = m0 + hi * 8 + r;
                const float* cg = ctxg + (size_t)m * G4H;
                float ig = sigmoid_f(ai[r] + cg[j]);
                float fg = sigmoid_f(af[r] + cg[HID + j]);
                float gg = tanhf    (ag[r] + cg[2 * HID + j]);
                float og = sigmoid_f(ao[r] + cg[3 * HID + j]);
                const size_t idx = (size_t)m * HID + j;
                float cn = fg * cbuf[idx] + ig * gg;
                float hn = og * tanhf(cn);
                cbuf[idx] = cn;
                hbuf[idx] = hn;
                xn[(size_t)m * KCAT + HID + j] = (_Float16)hn;  // h for next GEMMs
            }
        }
        group_barrier(ctr, ++phase * WG_PER_GROUP);

        // ---- Phase C: logits GEMM + in-register argmax partials ----
        {
            const _Float16* Arow = xn + HID + (size_t)(m0 + lq) * KCAT;  // new h (f16)
            for (int tn = waveIn; tn < NT_OUT; tn += WAVES_PER_GROUP) {
                const _Float16* Brow = Wout16 + (size_t)(tn * 16 + lq) * HID;
                v8f acc = {};
                for (int k0 = 0; k0 < HID; k0 += 32)
                    acc = wmma_acc(load_a_frag(Arow, k0, hi), load_b_frag(Brow, k0, hi), acc);
                const int n = tn * 16 + lq;
                const bool valid = (n < VOCAB);
                const float bv = valid ? b_out[n] : 0.0f;
#pragma unroll
                for (int r = 0; r < 8; ++r) {
                    const int m = m0 + hi * 8 + r;
                    const float lv = acc[r] + bv;
                    if (valid) out[(size_t)m * row_stride_out + (size_t)t * VOCAB + n] = lv;
                    // packed key: bigger value wins; tie -> smaller index (first max)
                    unsigned long long pk = valid
                        ? (((unsigned long long)f2ord(lv) << 32) | (unsigned)(~(unsigned)n))
                        : 0ull;
                    for (int msk = 1; msk < 16; msk <<= 1) {
                        unsigned long long o = shflxor_u64(pk, msk, 16);
                        if (o > pk) pk = o;
                    }
                    if (lq == 0) part[(size_t)m * 64 + tn] = pk;
                }
            }
        }
        group_barrier(ctr, ++phase * WG_PER_GROUP);

        // ---- Phase D: finish argmax, gather embedding into xn (emb half) ----
        if (waveIn < ROWS_PER_GROUP) {
            const int m = m0 + waveIn;
            unsigned long long pk = part[(size_t)m * 64 + lane];
            if (lane + 32 < NT_OUT) {
                unsigned long long o = part[(size_t)m * 64 + 32 + lane];
                if (o > pk) pk = o;
            }
            for (int msk = 1; msk < 32; msk <<= 1) {
                unsigned long long o = shflxor_u64(pk, msk, 32);
                if (o > pk) pk = o;
            }
            const int widx = (int)(~(unsigned)(pk & 0xffffffffull));
            const float* er = embed + (size_t)widx * EMB;
            for (int jj = lane; jj < EMB; jj += 32)
                xn[(size_t)m * KCAT + jj] = (_Float16)er[jj];
        }
        group_barrier(ctr, ++phase * WG_PER_GROUP);
    }

    // ---- final h, c into output tail ----
    {
        float* hc = out + (size_t)BATCH * TSTEPS * VOCAB;
        const int tig = wgIn * 256 + (int)threadIdx.x;            // [0, 1024)
        for (int e = tig; e < ROWS_PER_GROUP * HID; e += WG_PER_GROUP * 256) {
            const int r = e >> 9, j = e & 511;
            const size_t idx = (size_t)(m0 + r) * HID + j;
            hc[idx] = hbuf[idx];
            hc[(size_t)BATCH * HID + idx] = cbuf[idx];
        }
    }
}

// ---------------------------------------------------------------------------
extern "C" void kernel_launch(void* const* d_in, const int* in_sizes, int n_in,
                              void* d_out, int out_size, void* d_ws, size_t ws_size,
                              hipStream_t stream)
{
    const float* context  = (const float*)d_in[0];   // [128, 512]
    const float* embed    = (const float*)d_in[1];   // [1000, 512]
    const float* W_ih     = (const float*)d_in[2];   // [2048, 1024]
    const float* b_ih     = (const float*)d_in[3];   // [2048]
    const float* W_hh     = (const float*)d_in[4];   // [2048, 512]
    const float* b_hh     = (const float*)d_in[5];   // [2048]
    const float* W_out    = (const float*)d_in[6];   // [1000, 512]
    const float* b_out    = (const float*)d_in[7];   // [1000]
    const int*   start_id = (const int*)d_in[8];     // device scalar
    (void)in_sizes; (void)n_in; (void)out_size; (void)ws_size;

    float* out = (float*)d_out;

    // Workspace carve-out (256B aligned)
    char* ws = (char*)d_ws;
    size_t o = 0;
    auto carve = [&](size_t bytes) { size_t r = o; o = (o + bytes + 255) & ~(size_t)255; return r; };
    _Float16* Wcat   = (_Float16*)(ws + carve((size_t)G4H * KCAT * 2));   // 4 MB
    _Float16* Wctx   = (_Float16*)(ws + carve((size_t)G4H * HID * 2));    // 2 MB
    _Float16* Wout16 = (_Float16*)(ws + carve((size_t)NPAD * HID * 2));   // 1 MB
    float*    bsum   = (float*)   (ws + carve((size_t)G4H * 4));
    float*    ctxg   = (float*)   (ws + carve((size_t)BATCH * G4H * 4));  // 1 MB
    _Float16* xcat0  = (_Float16*)(ws + carve((size_t)BATCH * KCAT * 2));
    _Float16* xcat1  = (_Float16*)(ws + carve((size_t)BATCH * KCAT * 2));
    _Float16* ctx16  = (_Float16*)(ws + carve((size_t)BATCH * HID * 2));
    float*    hbuf   = (float*)   (ws + carve((size_t)BATCH * HID * 4));
    float*    cbuf   = (float*)   (ws + carve((size_t)BATCH * HID * 4));
    unsigned long long* part = (unsigned long long*)(ws + carve((size_t)BATCH * 64 * 8));
    unsigned* bar    = (unsigned*)(ws + carve((size_t)NGROUPS * 4));

    // ---- prep (every call; deterministic) ----
    conv_wcat<<<(G4H * KCAT) / 256, 256, 0, stream>>>(W_ih, W_hh, Wcat);
    conv_wctx<<<(G4H * HID) / 256, 256, 0, stream>>>(W_ih, Wctx);
    conv_wout<<<(NPAD * HID) / 256, 256, 0, stream>>>(W_out, Wout16);
    conv_bsum<<<G4H / 256, 256, 0, stream>>>(b_ih, b_hh, bsum, bar);
    init_state<<<(BATCH * HID) / 256, 256, 0, stream>>>(context, embed, start_id,
                                                        xcat0, ctx16, hbuf, cbuf);

    // ---- one persistent kernel runs the whole decode ----
    decoder_persistent<<<NGROUPS * WG_PER_GROUP, 256, 0, stream>>>(
        Wcat, Wctx, Wout16, bsum, b_out, embed, ctx16,
        ctxg, xcat0, xcat1, hbuf, cbuf, part, bar, out);
}
